// DeformableConv1d_60885456388530
// MI455X (gfx1250) — matrix operand order repaired and checked by reading, not compiled, not run on hardware
//
#include <hip/hip_runtime.h>

typedef unsigned int   u32;
typedef unsigned short u16;
typedef __attribute__((ext_vector_type(16))) _Float16 v16h;
typedef __attribute__((ext_vector_type(8)))  float    v8f;

union Frag { uint4 q[2]; v16h v; };

// ---------- f16 helpers ----------
__device__ __forceinline__ u16 f32_to_f16b(float f) {        // bits of f16
    return __builtin_bit_cast(u16, (_Float16)f);
}
__device__ __forceinline__ float f16lo(u32 u) {              // low half -> f32
    return (float)__builtin_bit_cast(_Float16, (u16)(u & 0xFFFFu));
}
__device__ __forceinline__ float f16hi(u32 u) {              // high half -> f32
    return (float)__builtin_bit_cast(_Float16, (u16)(u >> 16));
}
__device__ __forceinline__ u32 pack_pk(float a, float b) {   // v_cvt_pk_rtz_f16_f32
    return __builtin_bit_cast(u32, __builtin_amdgcn_cvt_pkrtz(a, b));
}

// Problem constants
#define NB   8
#define CIN  512
#define LEN  4096
#define LP   4098          // padded length
#define KTOT 1536          // CIN*KW
#define COUT 512

// =====================================================================
// Kernel 1: offset conv -> li / alpha tables  [n][k][l]
// block = 256 (8 waves). Block handles (n, 32 consecutive l).
// wave w covers channels [w*64, w*64+64); lane = l offset (coalesced x reads).
// =====================================================================
__global__ __launch_bounds__(256) void offsets_kernel(
    const float* __restrict__ x, const float* __restrict__ w_off,
    const float* __restrict__ b_off, int* __restrict__ Li, float* __restrict__ Al)
{
    __shared__ float part[8][3][32];
    int tid  = threadIdx.x;
    int n    = blockIdx.x >> 7;            // 128 l-tiles per batch
    int l0   = (blockIdx.x & 127) * 32;
    int wv   = tid >> 5, lane = tid & 31;
    int l    = l0 + lane;

    float a0 = 0.f, a1 = 0.f, a2 = 0.f;
    int cbeg = wv * 64;
    for (int c = cbeg; c < cbeg + 64; ++c) {
        const float* xr = x + ((size_t)n * CIN + c) * LEN;
        float xm = (l >= 1)       ? xr[l - 1] : 0.f;
        float x0 = xr[l];
        float xp = (l <= LEN - 2) ? xr[l + 1] : 0.f;
        const float* w0 = w_off + (size_t)c * 3;               // out-channel 0 (k=0)
        const float* w1 = w_off + 2 * (size_t)KTOT + c * 3;    // out-channel 2 (k=1)
        const float* w2 = w_off + 4 * (size_t)KTOT + c * 3;    // out-channel 4 (k=2)
        a0 += w0[0] * xm + w0[1] * x0 + w0[2] * xp;
        a1 += w1[0] * xm + w1[1] * x0 + w1[2] * xp;
        a2 += w2[0] * xm + w2[1] * x0 + w2[2] * xp;
    }
    part[wv][0][lane] = a0;
    part[wv][1][lane] = a1;
    part[wv][2][lane] = a2;
    __syncthreads();

    if (tid < 96) {
        int k = tid >> 5;                  // lane == tid & 31 == l offset
        float s = 0.f;
        #pragma unroll
        for (int w = 0; w < 8; ++w) s += part[w][k][lane];
        s += b_off[2 * k];
        float pos = (float)(l + 1) + s;
        pos = fminf(fmaxf(pos, 0.f), (float)(LP - 1));
        float lf = floorf(pos);
        int   li = (int)lf;
        size_t o = ((size_t)n * 3 + k) * LEN + l;
        Li[o] = li;
        Al[o] = pos - lf;
    }
}

// =====================================================================
// Kernel 2: x [n][c][l] f32  ->  Xp [n][1+l][c] f16 (padded transpose)
// =====================================================================
__global__ __launch_bounds__(256) void transpose_kernel(
    const float* __restrict__ x, u16* __restrict__ Xp)
{
    __shared__ float tile[32][33];
    int n = blockIdx.z, l0 = blockIdx.x * 32, c0 = blockIdx.y * 32;
    int lane = threadIdx.x & 31, row = threadIdx.x >> 5;
    #pragma unroll
    for (int r = 0; r < 4; ++r) {
        int ci = row + r * 8;
        tile[ci][lane] = x[((size_t)n * CIN + c0 + ci) * LEN + l0 + lane];
    }
    __syncthreads();
    #pragma unroll
    for (int r = 0; r < 4; ++r) {
        int lr = row + r * 8;
        Xp[((size_t)n * LP + 1 + l0 + lr) * CIN + c0 + lane] = f32_to_f16b(tile[lane][lr]);
    }
}

__global__ void pad_kernel(u16* __restrict__ Xp)
{
    int idx = blockIdx.x * 256 + threadIdx.x;   // 8 * 2 * 512
    if (idx >= NB * 2 * CIN) return;
    int n = idx >> 10, r = (idx >> 9) & 1, c = idx & 511;
    int row = r ? (LP - 1) : 0;
    Xp[((size_t)n * LP + row) * CIN + c] = 0;
}

__global__ void wconv_kernel(const float* __restrict__ w, u16* __restrict__ Wb)
{
    int idx = blockIdx.x * 256 + threadIdx.x;
    if (idx < COUT * KTOT) Wb[idx] = f32_to_f16b(w[idx]);
}

// =====================================================================
// Kernel 3: fused deformable-gather + GEMM via WMMA f16
//   Out[o, col] = b[o] + sum_i W[o,i] * lerp(Xp[n, li(i), c], Xp[n, ri(i), c])
//   col = n*4096 + lhi*512 + c ; l(i) = (i&511)*8 + lhi ; tap = i>>9
// WG tile 128(M) x 256(N); 8 waves 2x4; wave tile 64x64; K-step 64.
// =====================================================================
#define KSTEP 64
#define ASTR  72   // LDS row stride in u16 (144B: 16B aligned, conflict-reduced)
#define BSTR  72

__global__ __launch_bounds__(256) void gemm_kernel(
    const u16* __restrict__ Wb, const u16* __restrict__ Xp,
    const int* __restrict__ Li, const float* __restrict__ Al,
    const float* __restrict__ bias, float* __restrict__ out)
{
    __shared__ u16   As[128 * ASTR];
    __shared__ u16   Bs[256 * BSTR];
    __shared__ int   li_all[KTOT];
    __shared__ float al_all[KTOT];
    __shared__ float biasS[128];

    int tid  = threadIdx.x;
    int col0 = blockIdx.x * 256;
    int m0   = blockIdx.y * 128;
    int n    = col0 >> 12;
    int rem  = col0 & 4095;
    int lhi  = rem >> 9;
    int c0   = rem & 511;

    // Stage li/alpha for the whole K range of this (n, lhi)
    for (int i = tid; i < KTOT; i += 256) {
        int ktap = i >> 9, ilo = i & 511;
        int gl = ilo * 8 + lhi;
        size_t t = ((size_t)n * 3 + ktap) * LEN + gl;
        li_all[i] = Li[t];
        al_all[i] = Al[t];
    }
    if (tid < 128) biasS[tid] = bias[m0 + tid];

    v8f acc[4][4];
    {
        v8f z = {0.f, 0.f, 0.f, 0.f, 0.f, 0.f, 0.f, 0.f};
        #pragma unroll
        for (int mf = 0; mf < 4; ++mf)
            #pragma unroll
            for (int nf = 0; nf < 4; ++nf) acc[mf][nf] = z;
    }

    int wv = tid >> 5, lane = tid & 31;
    int wm = wv >> 2, wn = wv & 3;
    int halfhi = (lane >= 16) ? 1 : 0;
    int lr = lane & 15;
    int koffA = halfhi ? 16 : 0;    // u16 offset within 32-K subtile: K 0..15 / 16..31
    int koffB = halfhi ? 16 : 0;    // u16 offset within 32-K subtile: K 0..15 / 16..31

    __syncthreads();   // tables ready; also orders against first staging

    for (int ks = 0; ks < KTOT / KSTEP; ++ks) {
        int k0 = ks * KSTEP;
        // ---- stage A tile (128 x 64 f16 weights) ----
        {
            int r = tid >> 1, ko = (tid & 1) * 32;
            const uint4* src = (const uint4*)(Wb + (size_t)(m0 + r) * KTOT + k0 + ko);
            uint4* dst = (uint4*)(&As[r * ASTR + ko]);
            dst[0] = src[0];
            dst[1] = src[1];
            dst[2] = src[2];
            dst[3] = src[3];
        }
        // ---- build B tile (64 x 256): gather + lerp, f16 pack ----
        {
            int cp = tid & 127;         // column pair
            int kh = tid >> 7;          // which half of the 64 K values
            int cA = c0 + 2 * cp;       // even global channel
            const u16* xcol = Xp + (size_t)n * LP * CIN + cA;
            u32 outE[16], outO[16];
            #pragma unroll
            for (int t2 = 0; t2 < 16; ++t2) {
                float ve[2], vo[2];
                #pragma unroll
                for (int s = 0; s < 2; ++s) {
                    int kk = kh * 32 + t2 * 2 + s;
                    int li = li_all[k0 + kk];
                    float a = al_all[k0 + kk];
                    int ri = li + 1; if (ri > LP - 1) ri = LP - 1;
                    u32 ul = *(const u32*)(xcol + (size_t)li * CIN);
                    u32 ur = *(const u32*)(xcol + (size_t)ri * CIN);
                    float xl0 = f16lo(ul), xl1 = f16hi(ul);
                    float xr0 = f16lo(ur), xr1 = f16hi(ur);
                    ve[s] = xl0 + a * (xr0 - xl0);
                    vo[s] = xl1 + a * (xr1 - xl1);
                }
                outE[t2] = pack_pk(ve[0], ve[1]);
                outO[t2] = pack_pk(vo[0], vo[1]);
            }
            u32* rowE = (u32*)(&Bs[(2 * cp) * BSTR]) + kh * 16;
            u32* rowO = (u32*)(&Bs[(2 * cp + 1) * BSTR]) + kh * 16;
            #pragma unroll
            for (int t2 = 0; t2 < 16; ++t2) { rowE[t2] = outE[t2]; rowO[t2] = outO[t2]; }
        }
        __syncthreads();

        // ---- two 32-K subtiles: load frags (K = 16h + s, same for A and B) + 16 WMMA each ----
        #pragma unroll
        for (int kb = 0; kb < 2; ++kb) {
            Frag aF[4], bF[4];
            #pragma unroll
            for (int mf = 0; mf < 4; ++mf) {
                int rA = wm * 64 + mf * 16 + lr;
                const uint4* ap = (const uint4*)(&As[rA * ASTR + kb * 32 + koffA]);
                aF[mf].q[0] = ap[0];        // K koff..koff+7
                aF[mf].q[1] = ap[1];        // K koff+8..koff+15 (+16B)
            }
            #pragma unroll
            for (int nf = 0; nf < 4; ++nf) {
                int cL = wn * 64 + nf * 16 + lr;
                const uint4* bp = (const uint4*)(&Bs[cL * BSTR + kb * 32 + koffB]);
                bF[nf].q[0] = bp[0];
                bF[nf].q[1] = bp[1];
            }
            #pragma unroll
            for (int mf = 0; mf < 4; ++mf)
                #pragma unroll
                for (int nf = 0; nf < 4; ++nf)
                    acc[mf][nf] = __builtin_amdgcn_wmma_f32_16x16x32_f16(
                        false, aF[mf].v, false, bF[nf].v,
                        (short)0, acc[mf][nf], false, false);
        }
        __syncthreads();
    }

    // ---- epilogue: bias + fp32 store ----
    size_t outbase = (size_t)n * COUT * LEN + (size_t)lhi * 512;
    #pragma unroll
    for (int mf = 0; mf < 4; ++mf) {
        #pragma unroll
        for (int r = 0; r < 8; ++r) {
            int oL = wm * 64 + mf * 16 + r + (halfhi ? 8 : 0);
            float bb = biasS[oL];
            size_t orow = outbase + (size_t)(m0 + oL) * LEN;
            #pragma unroll
            for (int nf = 0; nf < 4; ++nf) {
                int cg = c0 + wn * 64 + nf * 16 + lr;
                out[orow + cg] = acc[mf][nf][r] + bb;
            }
        }
    }
}

// =====================================================================
extern "C" void kernel_launch(void* const* d_in, const int* in_sizes, int n_in,
                              void* d_out, int out_size, void* d_ws, size_t ws_size,
                              hipStream_t stream)
{
    (void)in_sizes; (void)n_in; (void)out_size; (void)ws_size;
    const float* x      = (const float*)d_in[0];
    const float* w_off  = (const float*)d_in[1];
    const float* b_off  = (const float*)d_in[2];
    const float* w_conv = (const float*)d_in[3];
    const float* b_conv = (const float*)d_in[4];
    float* out = (float*)d_out;

    char* ws = (char*)d_ws;
    int*   Li = (int*)(ws + 0);                       // 8*3*4096*4   = 393216 B
    float* Al = (float*)(ws + 393216);                // 393216 B
    u16*   Wb = (u16*)(ws + 786432);                  // 512*1536*2   = 1572864 B
    u16*   Xp = (u16*)(ws + 2359296);                 // 8*4098*512*2 = 33570816 B
                                                      // total ~35.9 MB

    offsets_kernel  <<<1024,              256, 0, stream>>>(x, w_off, b_off, Li, Al);
    transpose_kernel<<<dim3(128, 16, 8),  256, 0, stream>>>(x, Xp);
    pad_kernel      <<<32,                256, 0, stream>>>(Xp);
    wconv_kernel    <<<3072,              256, 0, stream>>>(w_conv, Wb);
    gemm_kernel     <<<dim3(128, 4),      256, 0, stream>>>(Wb, Xp, Li, Al, b_conv, out);
}
